// TrafficGATv2Improved_48661979463998
// MI455X (gfx1250) — compile-verified, matrix operand order checked by reading, not run on hardware
//
#include <hip/hip_runtime.h>
#include <math.h>

#define NNODES 20000
#define NEDGES 200000
#define HID 96
#define NH 6
#define HC 576   // NH*HID

typedef __attribute__((ext_vector_type(16))) __bf16 bf16x16;
typedef __attribute__((ext_vector_type(8)))  float  floatx8;

union BF16Frag { bf16x16 v; unsigned short u[16]; };

__device__ __forceinline__ unsigned short f2bf(float f) {
  unsigned int u = __builtin_bit_cast(unsigned int, f);
  u += 0x7FFFu + ((u >> 16) & 1u);           // round-to-nearest-even
  return (unsigned short)(u >> 16);
}

__device__ __forceinline__ void atomicMaxF(float* addr, float val) {
  unsigned int* ua = (unsigned int*)addr;
  unsigned int old = __builtin_bit_cast(unsigned int, *addr);
  while (true) {
    float f = __builtin_bit_cast(float, old);
    if (f >= val) return;
    unsigned int assumed = old;
    old = atomicCAS(ua, assumed, __builtin_bit_cast(unsigned int, val));
    if (old == assumed) return;
  }
}

// load a 16-element bf16 A-fragment: row-major, K-offset pattern {0..7, 16..23}
__device__ __forceinline__ void load_afrag(BF16Frag& a, const unsigned short* p) {
  __builtin_memcpy(&a.u[0], p, 16);        // K = base+0..7
  __builtin_memcpy(&a.u[8], p + 16, 16);   // K = base+16..23
}

// load a packed B fragment (32B contiguous per lane)
__device__ __forceinline__ void load_bfrag(BF16Frag& b, const unsigned short* p) {
  __builtin_memcpy(&b.u[0], p, 32);
}

// ---------------- utility kernels ----------------
__global__ void k_fill(float* __restrict__ p, float v, int n) {
  int i = blockIdx.x * blockDim.x + threadIdx.x;
  if (i < n) p[i] = v;
}

// pack f32 weight [K,Nc] -> bf16 WMMA-B fragment order:
// block (ks,ntile) of 512 ushorts; within block: lane*16 + j,
// kk = ks*32 + (lane>>4)*8 + (j&7) + ((j>>3)<<4), n = ntile*16 + (lane&15)
__global__ void k_pack_b(const float* __restrict__ B, unsigned short* __restrict__ P,
                         int K, int Nc) {
  int idx = blockIdx.x * blockDim.x + threadIdx.x;
  if (idx >= K * Nc) return;
  int j = idx & 15;
  int lane = (idx >> 4) & 31;
  int blk = idx >> 9;
  int ntiles = Nc >> 4;
  int ntile = blk % ntiles;
  int ks = blk / ntiles;
  int kk = ks * 32 + ((lane >> 4) << 3) + (j & 7) + ((j >> 3) << 4);
  int n = ntile * 16 + (lane & 15);
  P[idx] = f2bf(B[(long)kk * Nc + n]);
}

// ---------------- encoders: tiny-K GEMM + LN + ReLU ----------------
__global__ void k_encoder(const float* __restrict__ in, int K,
                          const float* __restrict__ w, const float* __restrict__ b,
                          const float* __restrict__ g, const float* __restrict__ beta,
                          float* __restrict__ outf, unsigned short* __restrict__ outb) {
  int n = blockIdx.x, c = threadIdx.x;          // blockDim = HID (96)
  float v = b[c];
  for (int k = 0; k < K; ++k) v += in[n * K + k] * w[k * HID + c];
  __shared__ float s1[HID], s2[HID];
  s1[c] = v; s2[c] = v * v;
  __syncthreads();
  float su = 0.f, sq = 0.f;
  for (int i = 0; i < HID; ++i) { su += s1[i]; sq += s2[i]; }
  float mu = su / HID;
  float var = sq / HID - mu * mu;
  float y = (v - mu) * rsqrtf(var + 1e-5f) * g[c] + beta[c];
  y = y > 0.f ? y : 0.f;
  if (outf) outf[(long)n * HID + c] = y;
  outb[(long)n * HID + c] = f2bf(y);
}

// ------- generic bf16 WMMA GEMM, packed-B, 2 M-tiles per wave: C = A*B + bias -------
__global__ void k_wmma_gemm_bias(const unsigned short* __restrict__ A,
                                 const unsigned short* __restrict__ BP, // packed
                                 const float* __restrict__ bias,
                                 float* __restrict__ C, int M, int K, int Nc) {
  int wave = threadIdx.x >> 5, lane = threadIdx.x & 31;
  int ntile = blockIdx.y * 4 + wave;
  if (ntile * 16 >= Nc) return;                  // wave-uniform
  int mbase = blockIdx.x * 32;                   // two 16-row tiles per wave
  int nbase = ntile * 16;
  int ln = lane & 15, half = lane >> 4, koff = half * 8;
  int arow0 = mbase + ln;
  int arow1 = arow0 + 16;
  int ntiles = Nc >> 4;
  floatx8 acc0 = {}, acc1 = {};
  for (int k0 = 0; k0 < K; k0 += 32) {
    BF16Frag a0, a1, b;
    load_afrag(a0, A + (long)arow0 * K + k0 + koff);
    load_afrag(a1, A + (long)arow1 * K + k0 + koff);
    load_bfrag(b, BP + ((size_t)((k0 >> 5) * ntiles + ntile) << 9) + lane * 16);
    acc0 = __builtin_amdgcn_wmma_f32_16x16x32_bf16(false, a0.v, false, b.v,
                                                   (short)0, acc0, false, false);
    acc1 = __builtin_amdgcn_wmma_f32_16x16x32_bf16(false, a1.v, false, b.v,
                                                   (short)0, acc1, false, false);
  }
  float bv = bias ? bias[nbase + ln] : 0.f;
#pragma unroll
  for (int r = 0; r < 8; ++r) {
    int m = mbase + r + half * 8;
    C[(long)m * Nc + nbase + ln] = acc0[r] + bv;
    C[(long)(m + 16) * Nc + nbase + ln] = acc1[r] + bv;
  }
}

// ---------------- fused edge GEMM + gather + leakyReLU + attw-dot + segmax ----------------
__global__ void k_gat_edge(const unsigned short* __restrict__ eb,   // [E,96] bf16
                           const unsigned short* __restrict__ WeP,  // packed [96,576]
                           const float* __restrict__ xl,            // [N,576]
                           const float* __restrict__ xr,            // [N,576]
                           const int* __restrict__ src, const int* __restrict__ dst,
                           const float* __restrict__ attw,          // flat [576]
                           float* __restrict__ logits,              // [E,6]
                           float* __restrict__ segmax) {            // [N,6]
  int wave = threadIdx.x >> 5, lane = threadIdx.x & 31;
  int et = blockIdx.x * 4 + wave;
  int ebase = et * 16;
  int ln = lane & 15, half = lane >> 4, koff = half * 8;
  int arow = ebase + ln;
  BF16Frag a[3];                                 // preload 3 K-steps of edge_emb tile
#pragma unroll
  for (int s = 0; s < 3; ++s)
    load_afrag(a[s], eb + (long)arow * HID + s * 32 + koff);
  int se[8], de[8];
#pragma unroll
  for (int r = 0; r < 8; ++r) {
    int e = ebase + r + half * 8;
    se[r] = src[e]; de[r] = dst[e];
  }
  for (int h = 0; h < NH; ++h) {
    float lp[8];
#pragma unroll
    for (int r = 0; r < 8; ++r) lp[r] = 0.f;
    for (int t = 0; t < 6; t += 2) {             // column-tile pairs within head
      int nt0 = h * 6 + t;
      int nt1 = nt0 + 1;
      floatx8 acc0 = {}, acc1 = {};
#pragma unroll
      for (int s = 0; s < 3; ++s) {              // two independent WMMA chains
        BF16Frag b0, b1;
        load_bfrag(b0, WeP + ((size_t)(s * 36 + nt0) << 9) + lane * 16);
        load_bfrag(b1, WeP + ((size_t)(s * 36 + nt1) << 9) + lane * 16);
        acc0 = __builtin_amdgcn_wmma_f32_16x16x32_bf16(false, a[s].v, false, b0.v,
                                                       (short)0, acc0, false, false);
        acc1 = __builtin_amdgcn_wmma_f32_16x16x32_bf16(false, a[s].v, false, b1.v,
                                                       (short)0, acc1, false, false);
      }
      int col0 = nt0 * 16 + ln;
      int col1 = nt1 * 16 + ln;
      float aw0 = attw[col0];
      float aw1 = attw[col1];
#pragma unroll
      for (int r = 0; r < 8; ++r) {
        float v0 = acc0[r] + xl[(long)se[r] * HC + col0] + xr[(long)de[r] * HC + col0];
        v0 = v0 > 0.f ? v0 : 0.2f * v0;          // leaky_relu(0.2)
        float v1 = acc1[r] + xl[(long)se[r] * HC + col1] + xr[(long)de[r] * HC + col1];
        v1 = v1 > 0.f ? v1 : 0.2f * v1;
        lp[r] += v0 * aw0 + v1 * aw1;
      }
    }
#pragma unroll
    for (int r = 0; r < 8; ++r) {                // reduce over 16 lanes within half-wave
      float v = lp[r];
      v += __shfl_xor(v, 1, 32);
      v += __shfl_xor(v, 2, 32);
      v += __shfl_xor(v, 4, 32);
      v += __shfl_xor(v, 8, 32);
      lp[r] = v;
    }
    if (ln == 0) {                               // lane0 -> rows 0..7, lane16 -> rows 8..15
#pragma unroll
      for (int r = 0; r < 8; ++r) {
        int e = ebase + r + half * 8;
        logits[e * NH + h] = lp[r];
        atomicMaxF(&segmax[de[r] * NH + h], lp[r]);
      }
    }
  }
}

// ---------------- segment softmax numerator + denominator ----------------
__global__ void k_softmax(const float* __restrict__ logits, const int* __restrict__ dst,
                          const float* __restrict__ segmax, float* __restrict__ exl,
                          float* __restrict__ den, int E) {
  int i = blockIdx.x * blockDim.x + threadIdx.x;
  if (i >= E * NH) return;
  int e = i / NH;
  int h = i - e * NH;
  int d = dst[e];
  float ex = expf(logits[i] - segmax[d * NH + h]);
  exl[i] = ex;
  atomicAdd(&den[d * NH + h], ex);
}

// ---------------- alpha-weighted scatter aggregation ----------------
__global__ void k_aggregate(const float* __restrict__ xl, const float* __restrict__ exl,
                            const float* __restrict__ den, const int* __restrict__ src,
                            const int* __restrict__ dst, float* __restrict__ acc, int E) {
  long i = (long)blockIdx.x * blockDim.x + threadIdx.x;   // E*144 threads, 4 channels each
  if (i >= (long)E * 144) return;
  int e = (int)(i / 144);
  int q = (int)(i - (long)e * 144);
  int c0 = q * 4;
  int h = c0 / HID;
  int s = src[e], d = dst[e];
  float alpha = exl[e * NH + h] / (den[d * NH + h] + 1e-16f);
  const float* xs = xl + (long)s * HC + c0;
  float* ad = acc + (long)d * HC + c0;
#pragma unroll
  for (int j = 0; j < 4; ++j) atomicAdd(ad + j, xs[j] * alpha);
}

// ---------------- head-mean + cbias (+ELU) + LN + residual ----------------
__global__ void k_node_update(const float* __restrict__ acc, const float* __restrict__ cbias,
                              const float* __restrict__ g, const float* __restrict__ beta,
                              float* __restrict__ xf, unsigned short* __restrict__ xb,
                              int do_elu) {
  int n = blockIdx.x, c = threadIdx.x;          // blockDim = 96
  float v = 0.f;
  for (int h = 0; h < NH; ++h) v += acc[(long)n * HC + h * HID + c];
  v = v * (1.f / NH) + cbias[c];
  if (do_elu) v = v > 0.f ? v : (expf(v) - 1.f);
  __shared__ float s1[HID], s2[HID];
  s1[c] = v; s2[c] = v * v;
  __syncthreads();
  float su = 0.f, sq = 0.f;
  for (int i = 0; i < HID; ++i) { su += s1[i]; sq += s2[i]; }
  float mu = su / HID, var = sq / HID - mu * mu;
  float y = (v - mu) * rsqrtf(var + 1e-5f) * g[c] + beta[c];
  float xn = xf[(long)n * HID + c] + y;
  xf[(long)n * HID + c] = xn;
  xb[(long)n * HID + c] = f2bf(xn);
}

// ---------------- predictor stage 1: fused gather-concat GEMM ----------------
__global__ void k_pred_p1(const unsigned short* __restrict__ xb,
                          const unsigned short* __restrict__ eb,
                          const int* __restrict__ src, const int* __restrict__ dst,
                          const unsigned short* __restrict__ p1P,  // packed [288,128]
                          const float* __restrict__ p1b,
                          float* __restrict__ h1) {                // [E,128]
  int wave = threadIdx.x >> 5, lane = threadIdx.x & 31;
  int ntile = blockIdx.y * 4 + wave;            // 0..7
  int ebase = blockIdx.x * 16;
  int ln = lane & 15, half = lane >> 4, koff = half * 8;
  int arow = ebase + ln;
  int sA = src[arow], dA = dst[arow];
  int nb = ntile * 16;
  floatx8 acc = {};
  for (int k0 = 0; k0 < 288; k0 += 32) {
    BF16Frag a, b;
#pragma unroll
    for (int j = 0; j < 16; ++j) {
      int kk = k0 + koff + (j & 7) + ((j >> 3) << 4);
      unsigned short av;
      if (kk < 96)       av = xb[(long)sA * HID + kk];
      else if (kk < 192) av = xb[(long)dA * HID + (kk - 96)];
      else               av = eb[(long)arow * HID + (kk - 192)];
      a.u[j] = av;
    }
    load_bfrag(b, p1P + ((size_t)((k0 >> 5) * 8 + ntile) << 9) + lane * 16);
    acc = __builtin_amdgcn_wmma_f32_16x16x32_bf16(false, a.v, false, b.v,
                                                  (short)0, acc, false, false);
  }
  float bv = p1b[nb + ln];
#pragma unroll
  for (int r = 0; r < 8; ++r) {
    int m = ebase + r + half * 8;
    h1[(long)m * 128 + nb + ln] = acc[r] + bv;
  }
}

// ---------------- row LN + ReLU -> bf16 (wave per row) ----------------
__global__ void k_row_ln_relu_bf16(const float* __restrict__ in, const float* __restrict__ g,
                                   const float* __restrict__ beta,
                                   unsigned short* __restrict__ out, int rows, int L) {
  int wave = threadIdx.x >> 5, lane = threadIdx.x & 31;
  int r = blockIdx.x * (blockDim.x >> 5) + wave;
  if (r >= rows) return;
  int per = L >> 5;                              // <= 4
  float v[4];
  float su = 0.f, sq = 0.f;
  for (int i = 0; i < per; ++i) {
    float t = in[(long)r * L + lane + 32 * i];
    v[i] = t; su += t; sq += t * t;
  }
  for (int m = 1; m < 32; m <<= 1) { su += __shfl_xor(su, m, 32); sq += __shfl_xor(sq, m, 32); }
  float mu = su / L, var = sq / L - mu * mu, rs = rsqrtf(var + 1e-5f);
  for (int i = 0; i < per; ++i) {
    int c = lane + 32 * i;
    float y = (v[i] - mu) * rs * g[c] + beta[c];
    y = y > 0.f ? y : 0.f;
    out[(long)r * L + c] = f2bf(y);
  }
}

// ---------------- final: LN(64) + ReLU + dot(p3) ----------------
__global__ void k_final(const float* __restrict__ in, const float* __restrict__ g,
                        const float* __restrict__ beta, const float* __restrict__ w3,
                        const float* __restrict__ b3, float* __restrict__ out, int rows) {
  int wave = threadIdx.x >> 5, lane = threadIdx.x & 31;
  int r = blockIdx.x * (blockDim.x >> 5) + wave;
  if (r >= rows) return;
  float v0 = in[(long)r * 64 + lane];
  float v1 = in[(long)r * 64 + lane + 32];
  float su = v0 + v1, sq = v0 * v0 + v1 * v1;
  for (int m = 1; m < 32; m <<= 1) { su += __shfl_xor(su, m, 32); sq += __shfl_xor(sq, m, 32); }
  float mu = su / 64.f, var = sq / 64.f - mu * mu, rs = rsqrtf(var + 1e-5f);
  float y0 = (v0 - mu) * rs * g[lane] + beta[lane];        y0 = y0 > 0.f ? y0 : 0.f;
  float y1 = (v1 - mu) * rs * g[lane + 32] + beta[lane + 32]; y1 = y1 > 0.f ? y1 : 0.f;
  float d = y0 * w3[lane] + y1 * w3[lane + 32];
  for (int m = 1; m < 32; m <<= 1) d += __shfl_xor(d, m, 32);
  if (lane == 0) out[r] = d + b3[0];
}

// =====================================================================
extern "C" void kernel_launch(void* const* d_in, const int* in_sizes, int n_in,
                              void* d_out, int out_size, void* d_ws, size_t ws_size,
                              hipStream_t stream) {
  (void)in_sizes; (void)n_in; (void)out_size; (void)ws_size;
  const float* x        = (const float*)d_in[0];
  const float* ea       = (const float*)d_in[1];
  const int*   eidx     = (const int*)d_in[2];
  const int*   src      = eidx;
  const int*   dst      = eidx + NEDGES;
  const float* ne_w = (const float*)d_in[3],  *ne_b = (const float*)d_in[4];
  const float* ne_g = (const float*)d_in[5],  *ne_be = (const float*)d_in[6];
  const float* ee_w = (const float*)d_in[7],  *ee_b = (const float*)d_in[8];
  const float* ee_g = (const float*)d_in[9],  *ee_be = (const float*)d_in[10];
  const float* Wl  = (const float*)d_in[11], *bl  = (const float*)d_in[12];
  const float* Wr  = (const float*)d_in[13], *br  = (const float*)d_in[14];
  const float* We  = (const float*)d_in[15];
  const float* attw = (const float*)d_in[16], *cbias = (const float*)d_in[17];
  const float* lng = (const float*)d_in[18], *lnb = (const float*)d_in[19];
  const float* p1w = (const float*)d_in[20], *p1b = (const float*)d_in[21];
  const float* p1g = (const float*)d_in[22], *p1be = (const float*)d_in[23];
  const float* p2w = (const float*)d_in[24], *p2b = (const float*)d_in[25];
  const float* p2g = (const float*)d_in[26], *p2be = (const float*)d_in[27];
  const float* p3w = (const float*)d_in[28], *p3b = (const float*)d_in[29];

  char* base = (char*)d_ws;
  size_t off = 0;
  auto alloc = [&](size_t bytes) -> void* {
    void* p = base + off;
    off = (off + bytes + 255) & ~(size_t)255;
    return p;
  };
  // persistent
  float*          xf   = (float*)alloc((size_t)NNODES * HID * 4);
  unsigned short* xb   = (unsigned short*)alloc((size_t)NNODES * HID * 2);
  unsigned short* ebuf = (unsigned short*)alloc((size_t)NEDGES * HID * 2);
  unsigned short* WlP  = (unsigned short*)alloc((size_t)3 * HID * HC * 2);
  unsigned short* WrP  = (unsigned short*)alloc((size_t)3 * HID * HC * 2);
  unsigned short* WeP  = (unsigned short*)alloc((size_t)3 * HID * HC * 2);
  unsigned short* p1P  = (unsigned short*)alloc((size_t)288 * 128 * 2);
  unsigned short* p2P  = (unsigned short*)alloc((size_t)128 * 64 * 2);
  float* segmax = (float*)alloc((size_t)NNODES * NH * 4);
  float* den    = (float*)alloc((size_t)NNODES * NH * 4);
  float* logits = (float*)alloc((size_t)NEDGES * NH * 4);
  float* exl    = (float*)alloc((size_t)NEDGES * NH * 4);
  // union region: (xl,xr,acc) during layers  /  (h1raw,h1b,h2raw) during predictor
  char* u = base + off;
  float* xl   = (float*)u;
  float* xr   = xl + (size_t)NNODES * HC;
  float* accb = xr + (size_t)NNODES * HC;
  float* h1raw = (float*)u;
  unsigned short* h1b = (unsigned short*)((char*)u + (size_t)NEDGES * 128 * 4);
  float* h2raw = (float*)((char*)h1b + (size_t)NEDGES * 128 * 2);

  const int T = 256;
  // pack weights into bf16 WMMA-B fragment order (deterministic, redone every call)
  for (int L = 0; L < 3; ++L) {
    size_t o = (size_t)L * HID * HC;
    k_pack_b<<<(HID * HC + T - 1) / T, T, 0, stream>>>(Wl + o, WlP + o, HID, HC);
    k_pack_b<<<(HID * HC + T - 1) / T, T, 0, stream>>>(Wr + o, WrP + o, HID, HC);
    k_pack_b<<<(HID * HC + T - 1) / T, T, 0, stream>>>(We + o, WeP + o, HID, HC);
  }
  k_pack_b<<<(288 * 128 + T - 1) / T, T, 0, stream>>>(p1w, p1P, 288, 128);
  k_pack_b<<<(128 * 64 + T - 1) / T, T, 0, stream>>>(p2w, p2P, 128, 64);

  // encoders
  k_encoder<<<NNODES, HID, 0, stream>>>(x, 4, ne_w, ne_b, ne_g, ne_be, xf, xb);
  k_encoder<<<NEDGES, HID, 0, stream>>>(ea, 3, ee_w, ee_b, ee_g, ee_be, nullptr, ebuf);

  for (int L = 0; L < 3; ++L) {
    k_fill<<<(NNODES * NH + T - 1) / T, T, 0, stream>>>(segmax, -3.0e38f, NNODES * NH);
    k_fill<<<(NNODES * NH + T - 1) / T, T, 0, stream>>>(den, 0.f, NNODES * NH);
    k_fill<<<(NNODES * HC + T - 1) / T, T, 0, stream>>>(accb, 0.f, NNODES * HC);

    dim3 g1(NNODES / 32, HC / 64);
    k_wmma_gemm_bias<<<g1, 128, 0, stream>>>(xb, WlP + (size_t)L * HID * HC, bl + L * HC,
                                             xl, NNODES, HID, HC);
    k_wmma_gemm_bias<<<g1, 128, 0, stream>>>(xb, WrP + (size_t)L * HID * HC, br + L * HC,
                                             xr, NNODES, HID, HC);
    k_gat_edge<<<NEDGES / 64, 128, 0, stream>>>(ebuf, WeP + (size_t)L * HID * HC, xl, xr,
                                                src, dst, attw + L * HC, logits, segmax);
    k_softmax<<<(NEDGES * NH + T - 1) / T, T, 0, stream>>>(logits, dst, segmax, exl, den, NEDGES);
    long tot = (long)NEDGES * 144;
    k_aggregate<<<(unsigned)((tot + T - 1) / T), T, 0, stream>>>(xl, exl, den, src, dst, accb, NEDGES);
    k_node_update<<<NNODES, HID, 0, stream>>>(accb, cbias + L * HID, lng + L * HID, lnb + L * HID,
                                              xf, xb, (L < 2) ? 1 : 0);
  }

  // predictor
  dim3 gp1(NEDGES / 16, 2);
  k_pred_p1<<<gp1, 128, 0, stream>>>(xb, ebuf, src, dst, p1P, p1b, h1raw);
  k_row_ln_relu_bf16<<<NEDGES / 8, 256, 0, stream>>>(h1raw, p1g, p1be, h1b, NEDGES, 128);
  dim3 gp2(NEDGES / 32, 1);
  k_wmma_gemm_bias<<<gp2, 128, 0, stream>>>(h1b, p2P, p2b, h2raw, NEDGES, 128, 64);
  k_final<<<NEDGES / 8, 256, 0, stream>>>(h2raw, p2g, p2be, p3w, p3b, (float*)d_out, NEDGES);
}
